// ARDecoder_63556926046594
// MI455X (gfx1250) — matrix operand order repaired and checked by reading, not compiled, not run on hardware
//
#include <hip/hip_runtime.h>
#include <hip/hip_bf16.h>

// ---------------------------------------------------------------------------
// CDNA5 (gfx1250, wave32) bf16-WMMA implementation of the ARDecoder reference.
// All GEMMs use v_wmma_f32_16x16x32_bf16 (f32 accumulate). Attention is a
// flash-style per-wave kernel (16-row query tile per wave) with online
// softmax; P is re-laid-out C-frag -> A-frag through a private LDS slab.
// V is stored transposed (B,H,A,S) so the P*V B-fragments are contiguous
// 32-byte loads. Weights are converted to bf16 once per call (~106 MB set,
// L2-resident on the 192 MB global L2).
// ---------------------------------------------------------------------------

typedef __bf16 bf16;
typedef __attribute__((ext_vector_type(16))) __bf16 v16bf;
typedef __attribute__((ext_vector_type(8)))  __bf16 v8bf;
typedef __attribute__((ext_vector_type(8)))  float  v8f;

constexpr int Bc = 2, BLKSc = 16, Kc = 16, Nc = 64, Dc = 1024;
constexpr int Hc = 16, Ac = 64, Fc = 2816, Lc = 4, WINc = 4;
constexpr int Ec = Kc + Nc;        // 80 tokens per block
constexpr int Sc = BLKSc * Ec;     // 1280 tokens per batch

__device__ __forceinline__ v8f wmma_bf16(v16bf a, v16bf b, v8f c) {
    return __builtin_amdgcn_wmma_f32_16x16x32_bf16(false, a, false, b,
                                                   (short)0, c, false, false);
}

// A-fragment: lane l -> row (l&15); elems 0..15 = K kof..kof+15
// (kof = 16*(l>=16)).  p must already point at [row, k0+kof].
__device__ __forceinline__ v16bf load_afrag(const bf16* p) {
    v8bf lo = *(const v8bf*)p;
    v8bf hi = *(const v8bf*)(p + 8);
    v16bf a;
#pragma unroll
    for (int j = 0; j < 8; ++j) { a[j] = lo[j]; a[j + 8] = hi[j]; }
    return a;
}

// B-fragment: lane l -> col (l&15); elems j = K (16*(l>=16))+j, contiguous 16.
__device__ __forceinline__ v16bf load_bfrag(const bf16* p) {
    return *(const v16bf*)p;
}

// ---------------------------------------------------------------------------
// elementwise / norm kernels
// ---------------------------------------------------------------------------
__global__ void cvt_bf16(const float* __restrict__ in, bf16* __restrict__ out,
                         size_t n) {
    size_t i = (size_t)blockIdx.x * blockDim.x + threadIdx.x;
    size_t stride = (size_t)gridDim.x * blockDim.x;
    for (; i < n; i += stride) out[i] = (bf16)in[i];
}

__global__ void build_maps(int* __restrict__ dmap, int* __restrict__ amap) {
    int j = blockIdx.x * 256 + threadIdx.x;
    if (j < Bc * BLKSc * Kc) {
        int b = j / (BLKSc * Kc); int rem = j % (BLKSc * Kc);
        int blk = rem / Kc; int r = rem % Kc;
        dmap[j] = b * Sc + blk * Ec + r;
    }
    if (j < Bc * BLKSc * Nc) {
        int b = j / (BLKSc * Nc); int rem = j % (BLKSc * Nc);
        int blk = rem / Nc; int r = rem % Nc;
        amap[j] = b * Sc + blk * Ec + Kc + r;
    }
}

__device__ __forceinline__ float block_sum256(float v) {
    __shared__ float red[8];
#pragma unroll
    for (int o = 16; o > 0; o >>= 1) v += __shfl_xor(v, o, 32);
    if ((threadIdx.x & 31) == 0) red[threadIdx.x >> 5] = v;
    __syncthreads();
    float t = (threadIdx.x < 8) ? red[threadIdx.x] : 0.f;
    if (threadIdx.x < 32) {
#pragma unroll
        for (int o = 4; o > 0; o >>= 1) t += __shfl_xor(t, o, 32);
    }
    if (threadIdx.x == 0) red[0] = t;
    __syncthreads();
    return red[0];
}

// rmsnorm(x+bias, w) -> bf16   (one 256-thread block per row of D=1024)
__global__ __launch_bounds__(256) void prep_norm(const float* __restrict__ src,
                                                 const float* __restrict__ bias,
                                                 const float* __restrict__ w,
                                                 bf16* __restrict__ out, int T) {
    int row = blockIdx.x;
    int r = row % T;
    const float* xr = src + (size_t)row * Dc;
    const float* br = bias + (size_t)r * Dc;
    float vals[4]; float s = 0.f;
#pragma unroll
    for (int j = 0; j < 4; ++j) {
        int i = threadIdx.x + j * 256;
        float v = xr[i] + br[i];
        vals[j] = v; s += v * v;
    }
    s = block_sum256(s);
    float inv = rsqrtf(s * (1.f / Dc) + 1e-5f);
#pragma unroll
    for (int j = 0; j < 4; ++j) {
        int i = threadIdx.x + j * 256;
        out[(size_t)row * Dc + i] = (bf16)(vals[j] * inv * w[i]);
    }
}

// rmsnorm(x, w) -> bf16
__global__ __launch_bounds__(256) void rmsnorm_x(const float* __restrict__ x,
                                                 const float* __restrict__ w,
                                                 bf16* __restrict__ out) {
    size_t row = blockIdx.x;
    const float* xr = x + row * (size_t)Dc;
    float vals[4]; float s = 0.f;
#pragma unroll
    for (int j = 0; j < 4; ++j) {
        int i = threadIdx.x + j * 256;
        float v = xr[i];
        vals[j] = v; s += v * v;
    }
    s = block_sum256(s);
    float inv = rsqrtf(s * (1.f / Dc) + 1e-5f);
#pragma unroll
    for (int j = 0; j < 4; ++j) {
        int i = threadIdx.x + j * 256;
        out[row * Dc + i] = (bf16)(vals[j] * inv * w[i]);
    }
}

// ---------------------------------------------------------------------------
// GEMM: C[M,N] (+=) A[M,Kd] * W[N,Kd]^T   (bf16 inputs, f32 accumulate)
// block = 256 threads = 8 waves (4x2); block tile 128(M) x 64(N);
// each wave: 32x32 via 2x2 WMMA accumulators. All dims are multiples.
// ---------------------------------------------------------------------------
__global__ __launch_bounds__(256) void gemm_bf16(
    const bf16* __restrict__ A, int lda,
    const bf16* __restrict__ W, int ldw,
    float* __restrict__ C, int ldc,
    int Kd, int beta, const int* __restrict__ rowmap) {
    int lane = threadIdx.x & 31;
    int wave = threadIdx.x >> 5;
    int m0 = blockIdx.x * 128 + (wave >> 1) * 32;
    int n0 = blockIdx.y * 64 + (wave & 1) * 32;
    int l15 = lane & 15;
    int hi = lane >> 4;
    int kofA = hi << 4;   // A-frag K offset (0 / 16)
    int kofB = hi << 4;   // B-frag K offset (0 / 16)

    const bf16* a0p = A + (size_t)(m0 + l15) * lda + kofA;
    const bf16* a1p = A + (size_t)(m0 + 16 + l15) * lda + kofA;
    const bf16* b0p = W + (size_t)(n0 + l15) * ldw + kofB;
    const bf16* b1p = W + (size_t)(n0 + 16 + l15) * ldw + kofB;

    v8f acc[2][2] = {};
    for (int k0 = 0; k0 < Kd; k0 += 32) {
        v16bf a0 = load_afrag(a0p + k0);
        v16bf a1 = load_afrag(a1p + k0);
        v16bf b0 = load_bfrag(b0p + k0);
        v16bf b1 = load_bfrag(b1p + k0);
        // pull next k-slab toward L2/L0 (global_prefetch_b8)
        __builtin_prefetch(a0p + k0 + 32, 0, 1);
        __builtin_prefetch(b0p + k0 + 32, 0, 1);
        acc[0][0] = wmma_bf16(a0, b0, acc[0][0]);
        acc[0][1] = wmma_bf16(a0, b1, acc[0][1]);
        acc[1][0] = wmma_bf16(a1, b0, acc[1][0]);
        acc[1][1] = wmma_bf16(a1, b1, acc[1][1]);
    }
    int rb = hi << 3;
#pragma unroll
    for (int i = 0; i < 2; ++i)
#pragma unroll
        for (int j = 0; j < 2; ++j)
#pragma unroll
            for (int r = 0; r < 8; ++r) {
                int m = m0 + i * 16 + rb + r;
                int n = n0 + j * 16 + l15;
                int row = rowmap ? rowmap[m] : m;
                size_t idx = (size_t)row * ldc + n;
                float v = acc[i][j][r];
                if (beta) C[idx] += v; else C[idx] = v;
            }
}

// ---------------------------------------------------------------------------
// q/k: l2norm + rope -> (B,H,S,A); v: cast -> transposed (B,H,A,S)
// ---------------------------------------------------------------------------
__device__ __forceinline__ void l2rope_store(const float* __restrict__ in,
                                             bf16* __restrict__ out, int s) {
    float x[Ac]; float ss = 0.f;
#pragma unroll
    for (int a = 0; a < Ac; ++a) { x[a] = in[a]; ss += x[a] * x[a]; }
    float inv = 1.f / fmaxf(sqrtf(ss), 1e-5f);
#pragma unroll
    for (int a = 0; a < Ac; ++a) x[a] *= inv;
    constexpr int half = Ac / 2;
#pragma unroll
    for (int j = 0; j < half; ++j) {
        // inv_freq = theta^(-2j/A); fr = s * inv_freq
        float fr = (float)s * __expf(-(2.f * j / (float)Ac) * 9.210340371976184f);
        float sn, c;
        __sincosf(fr, &sn, &c);
        out[j]        = (bf16)( x[j] * c + x[j + half] * sn);
        out[j + half] = (bf16)(-x[j] * sn + x[j + half] * c);
    }
}

__global__ __launch_bounds__(256) void qk_prep(const float* __restrict__ qkv,
                                               bf16* __restrict__ qb,
                                               bf16* __restrict__ kb,
                                               bf16* __restrict__ vbT) {
    int t = blockIdx.x * 256 + threadIdx.x;      // exactly B*S*H threads
    int h = t % Hc;
    int s = (t / Hc) % Sc;
    int b = t / (Hc * Sc);
    const float* basep = qkv + (size_t)(b * Sc + s) * (3 * Dc) + h * Ac;
    size_t orow = ((size_t)(b * Hc + h) * Sc + s) * (size_t)Ac;
    l2rope_store(basep,      qb + orow, s);
    l2rope_store(basep + Dc, kb + orow, s);
    // V transposed per head: vbT[((b*H+h)*A + a)*S + s]
    bf16* vt = vbT + ((size_t)(b * Hc + h) * Ac) * Sc + s;
#pragma unroll
    for (int a = 0; a < Ac; ++a) vt[(size_t)a * Sc] = (bf16)basep[2 * Dc + a];
}

// ---------------------------------------------------------------------------
// Flash attention: grid (B*H, S/16/8), block 256 = 8 waves.
// Each wave owns one 16-row q-tile; loops 32-key tiles with online softmax.
// Key range pruned to [(bq-WIN+1)*E, (bq+1)*E): outside it the mask is
// provably false (window / same-block-latent structure).
// ---------------------------------------------------------------------------
__global__ __launch_bounds__(256) void flash_attn(const bf16* __restrict__ qb,
                                                  const bf16* __restrict__ kb,
                                                  const bf16* __restrict__ vbT,
                                                  const int* __restrict__ doc,
                                                  bf16* __restrict__ ob) {
    __shared__ bf16 plds[8][16 * 32];           // per-wave P relayout slab
    int bh = blockIdx.x;
    int b = bh / Hc, h = bh % Hc;
    int wave = threadIdx.x >> 5, lane = threadIdx.x & 31;
    int qt = blockIdx.y * 8 + wave;             // grid sized exactly: no tail
    int qbase = qt * 16;
    int l15 = lane & 15;
    int hi = lane >> 4;
    int kofA = hi << 4;
    int rbase = hi << 3;

    const bf16* Q  = qb  + ((size_t)(b * Hc + h) * Sc) * Ac;
    const bf16* Km = kb  + ((size_t)(b * Hc + h) * Sc) * Ac;
    const bf16* Vt = vbT + ((size_t)(b * Hc + h) * Ac) * Sc;  // [A][S]

    // Q A-frags for head-dim chunks [0,32) and [32,64)
    v16bf qa[2];
    {
        const bf16* p = Q + (size_t)(qbase + l15) * Ac;
#pragma unroll
        for (int c = 0; c < 2; ++c)
#pragma unroll
            for (int j = 0; j < 8; ++j) {
                qa[c][j]     = p[c * 32 + kofA + j];
                qa[c][j + 8] = p[c * 32 + kofA + 8 + j];
            }
    }

    v8f accO[4] = {};
    float rm[8], rl[8];
#pragma unroll
    for (int r = 0; r < 8; ++r) { rm[r] = -3.0e38f; rl[r] = 0.f; }

    int bq = qbase / Ec;                 // 80 = 5*16 -> q-tile sits in one block
    int dq = doc[b * BLKSc + bq];
    bool tq8[8]; int sq8[8];
#pragma unroll
    for (int r = 0; r < 8; ++r) {
        sq8[r] = qbase + rbase + r;
        tq8[r] = (sq8[r] % Ec) >= Kc;
    }

    int kend = (bq + 1) * Ec;
    int ksb = bq - (WINc - 1); if (ksb < 0) ksb = 0;
    int k0 = (ksb * Ec) & ~31;
    const float scale = 0.125f;          // 1/sqrt(A)

    for (; k0 < kend; k0 += 32) {
        float sraw[2][8];
#pragma unroll
        for (int nt = 0; nt < 2; ++nt) {
            int col0 = k0 + nt * 16;
            const bf16* p = Km + (size_t)(col0 + l15) * Ac + (hi << 4);
            v16bf kf0 = *(const v16bf*)p;
            v16bf kf1 = *(const v16bf*)(p + 32);
            v8f s = {};
            s = wmma_bf16(qa[0], kf0, s);
            s = wmma_bf16(qa[1], kf1, s);
            // mask (per-column quantities are lane-invariant over r)
            int skc = col0 + l15;
            int bk = skc / Ec;
            int rk = skc - bk * Ec;
            bool tk = rk >= Kc;
            int dk  = doc[b * BLKSc + bk];
            int dkp = doc[b * BLKSc + (bk > 0 ? bk - 1 : 0)];
#pragma unroll
            for (int r = 0; r < 8; ++r) {
                bool tt = tq8[r] && tk && (sq8[r] >= skc) &&
                          (bq < bk + WINc) && (dq == dk);
                bool lm = ((tq8[r] && (dq == dkp)) || !tq8[r]) && !tk && (bq == bk);
                sraw[nt][r] = (tt || lm) ? s[r] * scale : -1.0e30f;
            }
        }
        // row max across the 16 columns in this lane-half
        float nm[8];
#pragma unroll
        for (int r = 0; r < 8; ++r)
            nm[r] = fmaxf(rm[r], fmaxf(sraw[0][r], sraw[1][r]));
#pragma unroll
        for (int off = 8; off > 0; off >>= 1)
#pragma unroll
            for (int r = 0; r < 8; ++r)
                nm[r] = fmaxf(nm[r], __shfl_xor(nm[r], off, 32));
        // rescale running state (O rows share the score-row mapping)
#pragma unroll
        for (int r = 0; r < 8; ++r) {
            float a_ = __expf(rm[r] - nm[r]);
            rl[r] *= a_; rm[r] = nm[r];
            accO[0][r] *= a_; accO[1][r] *= a_;
            accO[2][r] *= a_; accO[3][r] *= a_;
        }
        // p = exp(s - m); stash in LDS (C-frag layout -> row-major 16x32)
#pragma unroll
        for (int nt = 0; nt < 2; ++nt)
#pragma unroll
            for (int r = 0; r < 8; ++r) {
                float pv = __expf(sraw[nt][r] - rm[r]);
                sraw[nt][r] = pv;
                plds[wave][(rbase + r) * 32 + nt * 16 + l15] = (bf16)pv;
            }
        // row sums
#pragma unroll
        for (int r = 0; r < 8; ++r) {
            float t = sraw[0][r] + sraw[1][r];
#pragma unroll
            for (int off = 8; off > 0; off >>= 1) t += __shfl_xor(t, off, 32);
            rl[r] += t;
        }
        // P A-frag from LDS (16x32, contraction over the 32 keys)
        v16bf pa;
#pragma unroll
        for (int j = 0; j < 8; ++j) {
            pa[j]     = plds[wave][l15 * 32 + kofA + j];
            pa[j + 8] = plds[wave][l15 * 32 + kofA + 8 + j];
        }
        // O += P * V : V is transposed [A][S], so the B-frag (32 keys x 16 dims)
        // is a single contiguous 32B load per lane.
#pragma unroll
        for (int n = 0; n < 4; ++n) {
            const bf16* pv_ = Vt + (size_t)(n * 16 + l15) * Sc + k0 + (hi << 4);
            v16bf vf = *(const v16bf*)pv_;
            accO[n] = wmma_bf16(pa, vf, accO[n]);
        }
    }
    // normalize and emit bf16 (B,S,D) so the Wo GEMM can consume directly
#pragma unroll
    for (int n = 0; n < 4; ++n)
#pragma unroll
        for (int r = 0; r < 8; ++r) {
            float v = accO[n][r] / rl[r];
            ob[((size_t)b * Sc + sq8[r]) * Dc + h * Ac + n * 16 + l15] = (bf16)v;
        }
}

// ---------------------------------------------------------------------------
__global__ void swiglu(const float* __restrict__ u, bf16* __restrict__ g) {
    size_t total = (size_t)Bc * Sc * Fc;
    size_t i = (size_t)blockIdx.x * blockDim.x + threadIdx.x;
    size_t stride = (size_t)gridDim.x * blockDim.x;
    for (; i < total; i += stride) {
        size_t row = i / Fc, c = i % Fc;
        float a = u[row * (2 * Fc) + c];
        float bb = u[row * (2 * Fc) + Fc + c];
        g[i] = (bf16)((a / (1.f + __expf(-a))) * bb);
    }
}

__global__ void out_copy(const float* __restrict__ x, float* __restrict__ out) {
    size_t total = (size_t)Bc * BLKSc * Nc * Dc;
    size_t i = (size_t)blockIdx.x * blockDim.x + threadIdx.x;
    size_t stride = (size_t)gridDim.x * blockDim.x;
    for (; i < total; i += stride) {
        int d = (int)(i % Dc);
        size_t r = i / Dc;
        int n = (int)(r % Nc);
        size_t r2 = r / Nc;
        int blk = (int)(r2 % BLKSc);
        int b = (int)(r2 / BLKSc);
        out[i] = x[((size_t)b * Sc + blk * Ec + Kc + n) * Dc + d];
    }
}

// ---------------------------------------------------------------------------
extern "C" void kernel_launch(void* const* d_in, const int* in_sizes, int n_in,
                              void* d_out, int out_size, void* d_ws, size_t ws_size,
                              hipStream_t stream) {
    const float* x_de    = (const float*)d_in[0];
    const float* x_ar    = (const float*)d_in[1];
    const int*   doc     = (const int*)d_in[2];
    // d_in[3] = begin (unused by the reference)
    const float* Wqkv    = (const float*)d_in[4];
    const float* Wo      = (const float*)d_in[5];
    const float* Wup     = (const float*)d_in[6];
    const float* Wdn     = (const float*)d_in[7];
    const float* attn_nw = (const float*)d_in[8];
    const float* ffn_nw  = (const float*)d_in[9];
    const float* de_nw   = (const float*)d_in[10];
    const float* de_W    = (const float*)d_in[11];
    const float* ar_nw   = (const float*)d_in[12];
    const float* ar_W    = (const float*)d_in[13];
    const float* pos_emb = (const float*)d_in[14];
    const float* ar_bias = (const float*)d_in[15];
    float* outp = (float*)d_out;

    char* base = (char*)d_ws;
    size_t off = 0;
    auto carve = [&](size_t bytes) -> char* {
        char* p = base + off;
        off += (bytes + 255) & ~(size_t)255;
        return p;
    };
    const size_t BSD = (size_t)Bc * Sc * Dc;      // 2.62M elems
    bf16* wde  = (bf16*)carve((size_t)Dc * Dc * 2);
    bf16* war  = (bf16*)carve((size_t)Dc * Dc * 2);
    bf16* wqkv = (bf16*)carve((size_t)Lc * 3 * Dc * Dc * 2);
    bf16* wwo  = (bf16*)carve((size_t)Lc * Dc * Dc * 2);
    bf16* wupb = (bf16*)carve((size_t)Lc * 2 * Fc * Dc * 2);
    bf16* wdnb = (bf16*)carve((size_t)Lc * Dc * Fc * 2);
    bf16* nde  = (bf16*)carve((size_t)Bc * BLKSc * Kc * Dc * 2);
    bf16* nar  = (bf16*)carve((size_t)Bc * BLKSc * Nc * Dc * 2);
    float* x   = (float*)carve(BSD * 4);
    bf16* hb   = (bf16*)carve(BSD * 2);
    float* qkvb = (float*)carve((size_t)Bc * Sc * 3 * Dc * 4);
    bf16* qbp  = (bf16*)carve(BSD * 2);
    bf16* kbp  = (bf16*)carve(BSD * 2);
    bf16* vbpT = (bf16*)carve(BSD * 2);           // (B,H,A,S) transposed V
    bf16* obp  = (bf16*)carve(BSD * 2);
    float* ub  = (float*)carve((size_t)Bc * Sc * 2 * Fc * 4);
    bf16* gb   = (bf16*)carve((size_t)Bc * Sc * Fc * 2);
    int* dmap  = (int*)carve((size_t)Bc * BLKSc * Kc * 4);
    int* amap  = (int*)carve((size_t)Bc * BLKSc * Nc * 4);
    (void)in_sizes; (void)n_in; (void)out_size; (void)ws_size;

    // ---- one-time-per-call weight conversion (bf16 set fits in 192MB L2) ----
    cvt_bf16<<<512,  256, 0, stream>>>(de_W, wde, (size_t)Dc * Dc);
    cvt_bf16<<<512,  256, 0, stream>>>(ar_W, war, (size_t)Dc * Dc);
    cvt_bf16<<<2048, 256, 0, stream>>>(Wqkv, wqkv, (size_t)Lc * 3 * Dc * Dc);
    cvt_bf16<<<1024, 256, 0, stream>>>(Wo,   wwo,  (size_t)Lc * Dc * Dc);
    cvt_bf16<<<2048, 256, 0, stream>>>(Wup,  wupb, (size_t)Lc * 2 * Fc * Dc);
    cvt_bf16<<<2048, 256, 0, stream>>>(Wdn,  wdnb, (size_t)Lc * Dc * Fc);
    build_maps<<<8, 256, 0, stream>>>(dmap, amap);

    // ---- input projections (write the interleaved residual stream x) ----
    prep_norm<<<Bc * BLKSc * Kc, 256, 0, stream>>>(x_de, pos_emb, de_nw, nde, Kc);
    prep_norm<<<Bc * BLKSc * Nc, 256, 0, stream>>>(x_ar, ar_bias, ar_nw, nar, Nc);
    {
        dim3 g(Bc * BLKSc * Kc / 128, Dc / 64);
        gemm_bf16<<<g, 256, 0, stream>>>(nde, Dc, wde, Dc, x, Dc, Dc, 0, dmap);
    }
    {
        dim3 g(Bc * BLKSc * Nc / 128, Dc / 64);
        gemm_bf16<<<g, 256, 0, stream>>>(nar, Dc, war, Dc, x, Dc, Dc, 0, amap);
    }

    const int Mtok = Bc * Sc;   // 2560
    for (int i = 0; i < Lc; ++i) {
        // attention
        rmsnorm_x<<<Mtok, 256, 0, stream>>>(x, attn_nw + (size_t)i * Dc, hb);
        {
            dim3 g(Mtok / 128, (3 * Dc) / 64);
            gemm_bf16<<<g, 256, 0, stream>>>(hb, Dc, wqkv + (size_t)i * 3 * Dc * Dc,
                                             Dc, qkvb, 3 * Dc, Dc, 0, nullptr);
        }
        qk_prep<<<(Bc * Sc * Hc) / 256, 256, 0, stream>>>(qkvb, qbp, kbp, vbpT);
        {
            dim3 g(Bc * Hc, (Sc / 16) / 8);
            flash_attn<<<g, 256, 0, stream>>>(qbp, kbp, vbpT, doc, obp);
        }
        {
            dim3 g(Mtok / 128, Dc / 64);
            gemm_bf16<<<g, 256, 0, stream>>>(obp, Dc, wwo + (size_t)i * Dc * Dc,
                                             Dc, x, Dc, Dc, 1, nullptr);
        }
        // FFN
        rmsnorm_x<<<Mtok, 256, 0, stream>>>(x, ffn_nw + (size_t)i * Dc, hb);
        {
            dim3 g(Mtok / 128, (2 * Fc) / 64);
            gemm_bf16<<<g, 256, 0, stream>>>(hb, Dc, wupb + (size_t)i * 2 * Fc * Dc,
                                             Dc, ub, 2 * Fc, Dc, 0, nullptr);
        }
        swiglu<<<2048, 256, 0, stream>>>(ub, gb);
        {
            dim3 g(Mtok / 128, Dc / 64);
            gemm_bf16<<<g, 256, 0, stream>>>(gb, Fc, wdnb + (size_t)i * Dc * Fc,
                                             Fc, x, Dc, Fc, 1, nullptr);
        }
    }
    out_copy<<<2048, 256, 0, stream>>>(x, outp);
}
